// AST_GRU_60498909331657
// MI455X (gfx1250) — matrix operand order repaired and checked by reading, not compiled, not run on hardware
//
#include <hip/hip_runtime.h>
#include <hip/hip_bf16.h>
#include <stdint.h>

// ---------------------------------------------------------------------------
// DAG-GRU for MI455X (gfx1250, wave32, WMMA + async-to-LDS staging).
// Leveled-DAG schedule: node i is in level i/6250, edges go l -> l+1, so each
// node's GRU cell is evaluated exactly once at its level (8x less GEMM work
// than the reference's sweep-all formulation, bit-identical semantics).
// GEMMs: v_wmma_f32_16x16x32_f16, fp32 accumulate. Tiles are staged into LDS
// with global_load_async_to_lds_b128 (ASYNCcnt), double buffered.
// ---------------------------------------------------------------------------

typedef __attribute__((ext_vector_type(16))) _Float16 v16h;
typedef __attribute__((ext_vector_type(8)))  _Float16 v8h;
typedef __attribute__((ext_vector_type(8)))  float    v8f;

#define HID      256
#define NLEV     8
#define NNODES   50000
#define PERLEV   (NNODES / NLEV)     // 6250
#define GATES_W  (3 * HID * 2)       // 1536 = [gi(768) | gh(768)]

#define BM 64
#define BN 128
#define BK 32
#define LDSW 40                      // halfs per LDS row (32 + 8 pad, 16B-aligned rows)

// --- gfx1250 async copy: global -> LDS, 16B per lane, tracked by ASYNCcnt ---
__device__ __forceinline__ void async_ld_b128(uint32_t ldsAddr, uint32_t vOff, uint64_t sBase) {
    asm volatile("global_load_async_to_lds_b128 %0, %1, %2"
                 :: "v"(ldsAddr), "v"(vOff), "s"(sBase) : "memory");
}
__device__ __forceinline__ void wait_async0() {
    asm volatile("s_wait_asynccnt 0x0" ::: "memory");
}

// ---------------------------------------------------------------------------
__global__ void k_f32_to_f16(const float* __restrict__ src,
                             _Float16* __restrict__ dst, int n) {
    int i = blockIdx.x * blockDim.x + threadIdx.x;
    int stride = gridDim.x * blockDim.x;
    for (; i < n; i += stride) dst[i] = (_Float16)src[i];
}

__global__ void k_zero(float* __restrict__ p, int n) {
    int i = blockIdx.x * blockDim.x + threadIdx.x;
    int stride = gridDim.x * blockDim.x;
    for (; i < n; i += stride) p[i] = 0.0f;
}

// ---------------------------------------------------------------------------
// Tiled WMMA GEMM:  out[m][col] = sum_k A[m][k] * W[colW][k]  (+ bias[colW])
// N-tiles split in two banks: bank0 = (A0,W0,bias0), bank1 = (A1,W1,bias1),
// so gi = x@w_ih^T + b_ih and gh = agg@w_hh^T + b_hh run in a single launch.
// All operands are f16 in HBM; tiles staged via async-to-LDS, double buffered.
// ---------------------------------------------------------------------------
__global__ __launch_bounds__(256)
void k_gemm_wmma(const _Float16* __restrict__ A0, const _Float16* __restrict__ A1,
                 const _Float16* __restrict__ W0, const _Float16* __restrict__ W1,
                 const float* __restrict__ bias0, const float* __restrict__ bias1,
                 float* __restrict__ out, int ldOut,
                 int rowCount, int nTilesHalf)
{
    __shared__ _Float16 lsA[2][BM * LDSW];   //  5 KB x2
    __shared__ _Float16 lsB[2][BN * LDSW];   // 10 KB x2

    const int tid   = threadIdx.x;
    const int lane  = tid & 31;
    const int wave  = tid >> 5;
    const int waveM = wave >> 1;             // 0..3  (16-row strip)
    const int waveN = wave & 1;              // 0..1  (64-col strip)
    const int g     = lane >> 4;             // lane half-group
    const int ln    = lane & 15;

    const int blockRow = blockIdx.x * BM;                // local row offset
    const int tile0    = blockIdx.y * (BN / 16);         // first 16-wide N tile
    const int bank     = (tile0 >= nTilesHalf) ? 1 : 0;
    const _Float16* A    = bank ? A1    : A0;
    const _Float16* W    = bank ? W1    : W0;
    const float*    bias = bank ? bias1 : bias0;
    const int colW0 = blockIdx.y * BN - bank * (nTilesHalf * 16);

    // --- async staging assignments (256 threads) ---
    const int ar = tid >> 2;            // A row   0..63
    const int ac = (tid & 3) * 8;       // A cols: 8 halfs (16B) each
    const int br = tid >> 1;            // W row   0..127
    const int bc = (tid & 1) * 16;      // W cols: 16 halfs (2x16B) each

    int arow = blockRow + ar;
    if (arow >= rowCount) arow = rowCount - 1;       // clamp (dup rows, masked at store)

    const uint64_t aBase = (uint64_t)(uintptr_t)A;
    const uint64_t wBase = (uint64_t)(uintptr_t)(W + (size_t)colW0 * HID);
    const uint32_t aVoff = (uint32_t)((arow * HID + ac) * 2);
    const uint32_t bVoff = (uint32_t)((br * HID + bc) * 2);

    uint32_t aLds[2], bLds[2];
    #pragma unroll
    for (int b = 0; b < 2; ++b) {
        aLds[b] = (uint32_t)(uintptr_t)&lsA[b][ar * LDSW + ac];   // low 32 bits = LDS offset
        bLds[b] = (uint32_t)(uintptr_t)&lsB[b][br * LDSW + bc];
    }

    auto issue = [&](int kk) {
        const int buf = kk & 1;
        const uint32_t kByte = (uint32_t)(kk * BK * 2);
        async_ld_b128(aLds[buf],      aVoff + kByte,      aBase);
        async_ld_b128(bLds[buf],      bVoff + kByte,      wBase);
        async_ld_b128(bLds[buf] + 16, bVoff + kByte + 16, wBase);
    };

    v8f acc[4];
    #pragma unroll
    for (int t = 0; t < 4; ++t)
        #pragma unroll
        for (int i = 0; i < 8; ++i) acc[t][i] = 0.0f;

    issue(0);
    wait_async0();
    __syncthreads();

    const int am = waveM * 16 + ln;
    #pragma unroll
    for (int kk = 0; kk < HID / BK; ++kk) {
        if (kk < HID / BK - 1) issue(kk + 1);        // fill other buffer while computing

        // A fragment: 16x32 f16, k = 16*g + s (contiguous, matches B) (two ds_load_b128)
        union { v16h v; v8h h[2]; } afr;
        afr.h[0] = *(const v8h*)&lsA[kk & 1][am * LDSW + g * 16];
        afr.h[1] = *(const v8h*)&lsA[kk & 1][am * LDSW + g * 16 + 8];

        #pragma unroll
        for (int nt = 0; nt < 4; ++nt) {
            // B fragment: 32x16 f16 (k-major per lane column)
            union { v16h v; v8h h[2]; } bfr;
            const int bn = waveN * 64 + nt * 16 + ln;
            bfr.h[0] = *(const v8h*)&lsB[kk & 1][bn * LDSW + g * 16];
            bfr.h[1] = *(const v8h*)&lsB[kk & 1][bn * LDSW + g * 16 + 8];
            acc[nt] = __builtin_amdgcn_wmma_f32_16x16x32_f16(
                false, afr.v, false, bfr.v, (short)0, acc[nt], false, false);
        }
        if (kk < HID / BK - 1) wait_async0();        // next buffer fully landed in LDS
        __syncthreads();
    }

    // Epilogue: C/D layout -> VGPR r holds rows (r + 8*g), col = ln
    #pragma unroll
    for (int nt = 0; nt < 4; ++nt) {
        const int col  = blockIdx.y * BN + waveN * 64 + nt * 16 + ln;
        const int colW = colW0 + waveN * 64 + nt * 16 + ln;
        const float bv = bias ? bias[colW] : 0.0f;
        #pragma unroll
        for (int vr = 0; vr < 8; ++vr) {
            const int mrow = blockRow + waveM * 16 + g * 8 + vr;
            if (mrow < rowCount)
                out[(size_t)mrow * ldOut + col] = acc[nt][vr] + bv;
        }
    }
}

// ---------------------------------------------------------------------------
// GRU gate math for one level. gates[i][0..767] = x@w_ih^T + b_ih,
// gates[i][768..1535] = agg@w_hh^T + b_hh (biases fused in GEMM epilogue).
// hidden (== d_out) is updated in place.
// ---------------------------------------------------------------------------
__global__ void k_gates(const float* __restrict__ gates,
                        const float* __restrict__ agg,
                        float* __restrict__ hidden, int nodeBase)
{
    int idx = blockIdx.x * blockDim.x + threadIdx.x;
    if (idx >= PERLEV * HID) return;
    const int i = idx >> 8;          // local row
    const int t = idx & (HID - 1);   // feature
    const float* gr = gates + (size_t)i * GATES_W;

    const float r  = 1.0f / (1.0f + __expf(-(gr[t]       + gr[768 + t])));
    const float z  = 1.0f / (1.0f + __expf(-(gr[256 + t] + gr[1024 + t])));
    const float nn = tanhf(gr[512 + t] + r * gr[1280 + t]);

    const int node = nodeBase + i;
    const float h  = agg[(size_t)node * HID + t];
    hidden[(size_t)node * HID + t] = (1.0f - z) * nn + z * h;
}

// ---------------------------------------------------------------------------
// Scatter hidden[src] into agg[dst] for edges whose src is in [lo, hi).
// 256 threads = one feature column each; frontier rows are L2-resident.
// ---------------------------------------------------------------------------
__global__ void k_scatter(const int* __restrict__ src, const int* __restrict__ dst,
                          const float* __restrict__ hidden, float* __restrict__ agg,
                          int nEdges, int lo, int hi)
{
    const int t = threadIdx.x;       // 0..255
    for (int e = blockIdx.x; e < nEdges; e += gridDim.x) {
        __builtin_prefetch(&src[e + gridDim.x], 0, 1);   // global_prefetch_b8
        const int s = src[e];
        if (s >= lo && s < hi) {
            const int d = dst[e];
            atomicAdd(&agg[(size_t)d * HID + t], hidden[(size_t)s * HID + t]);
        }
    }
}

// ---------------------------------------------------------------------------
extern "C" void kernel_launch(void* const* d_in, const int* in_sizes, int n_in,
                              void* d_out, int out_size, void* d_ws, size_t ws_size,
                              hipStream_t stream) {
    const float* V       = (const float*)d_in[0];
    const int*   E       = (const int*)  d_in[1];
    const float* dense_w = (const float*)d_in[2];
    const float* dense_b = (const float*)d_in[3];
    const float* w_ih    = (const float*)d_in[4];   // [2][768][256]
    const float* w_hh    = (const float*)d_in[5];
    const float* b_ih    = (const float*)d_in[6];   // [2][768]
    const float* b_hh    = (const float*)d_in[7];
    float* hidden = (float*)d_out;                  // in-place hidden state
    const int nE = in_sizes[1] / 2;

    // --- workspace carve-up ---
    auto alignup = [](size_t x) { return (x + 255) & ~(size_t)255; };
    char* ws = (char*)d_ws;
    _Float16* wd16  = (_Float16*)ws; ws += alignup((size_t)HID * HID * 2);
    _Float16* wih16 = (_Float16*)ws; ws += alignup((size_t)2 * 3 * HID * HID * 2);
    _Float16* whh16 = (_Float16*)ws; ws += alignup((size_t)2 * 3 * HID * HID * 2);
    _Float16* x16   = (_Float16*)ws; ws += alignup((size_t)NNODES * HID * 2);  // V16 at init, level x16 after
    _Float16* agg16 = (_Float16*)ws; ws += alignup((size_t)PERLEV * HID * 2);
    float* agg   = (float*)ws;       ws += alignup((size_t)NNODES * HID * 4);
    float* gates = (float*)ws;       ws += alignup((size_t)PERLEV * GATES_W * 4);

    // --- one-time f16 conversions (weights + V) ---
    k_f32_to_f16<<<128, 256, 0, stream>>>(dense_w, wd16, HID * HID);
    k_f32_to_f16<<<512, 256, 0, stream>>>(w_ih, wih16, 2 * 3 * HID * HID);
    k_f32_to_f16<<<512, 256, 0, stream>>>(w_hh, whh16, 2 * 3 * HID * HID);
    k_f32_to_f16<<<1024, 256, 0, stream>>>(V, x16, NNODES * HID);

    // --- init: hidden = V @ dense_w^T + dense_b ---
    {
        dim3 grid((NNODES + BM - 1) / BM, HID / BN);   // (782, 2)
        k_gemm_wmma<<<grid, 256, 0, stream>>>(
            x16, x16, wd16, wd16, dense_b, dense_b,
            hidden, HID, /*rowCount=*/NNODES, /*nTilesHalf=*/HID / 16);
    }

    // --- two DAG-GRU layers, processed level by level ---
    for (int l = 0; l < 2; ++l) {
        k_zero<<<1024, 256, 0, stream>>>(agg, NNODES * HID);
        const _Float16* Wih = wih16 + (size_t)l * 3 * HID * HID;
        const _Float16* Whh = whh16 + (size_t)l * 3 * HID * HID;
        const float*    Bih = b_ih + (size_t)l * 3 * HID;
        const float*    Bhh = b_hh + (size_t)l * 3 * HID;

        for (int lv = 0; lv < NLEV; ++lv) {
            const int base = lv * PERLEV;

            // level-local f16 copies of x and agg (A operands for the GEMM)
            k_f32_to_f16<<<512, 256, 0, stream>>>(hidden + (size_t)base * HID, x16,   PERLEV * HID);
            k_f32_to_f16<<<512, 256, 0, stream>>>(agg    + (size_t)base * HID, agg16, PERLEV * HID);

            // gi | gh for this level's 6250 nodes in one fused launch
            dim3 grid((PERLEV + BM - 1) / BM, GATES_W / BN);   // (98, 12)
            k_gemm_wmma<<<grid, 256, 0, stream>>>(
                x16, agg16, Wih, Whh, Bih, Bhh,
                gates, GATES_W, PERLEV, /*nTilesHalf=*/48);

            k_gates<<<(PERLEV * HID + 255) / 256, 256, 0, stream>>>(
                gates, agg, hidden, base);

            if (lv < NLEV - 1)
                k_scatter<<<2048, 256, 0, stream>>>(
                    E, E + nE, hidden, agg, nE, base, base + PERLEV);
        }
    }
}
